// Seq2Seq_25245817766425
// MI455X (gfx1250) — compile-verified, matrix operand order checked by reading, not run on hardware
//
#include <hip/hip_runtime.h>
#include <hip/hip_bf16.h>

// ---------------------------------------------------------------------------
// Seq2Seq (encoder LSTM + attention decoder) for MI455X / gfx1250, wave32.
// All GEMMs -> v_wmma_f32_16x16x32_f16 chains with bias / input-projection
// folded into augmented K (zero-init accumulators, no scalar prologue).
// Tile loops kept rolled to avoid scratch spills of weight streams.
// ---------------------------------------------------------------------------

#define B_   256
#define S_   512
#define T_   64
#define IN_  12
#define H_   256
#define G_   1024   // 4*H

#define KE_  288    // encoder / attn / lin1 augmented K: [256 | 12 | 1 | pad] -> 9x32
#define KD_  544    // decoder LSTM augmented K: [12 | 256 | 256 | 1 | pad] -> 17x32

typedef __attribute__((ext_vector_type(16))) _Float16 v16h;
typedef __attribute__((ext_vector_type(8)))  _Float16 v8h;
typedef __attribute__((ext_vector_type(8)))  float    v8f;

#if __has_builtin(__builtin_amdgcn_tanhf)
#define TANH_(x) __builtin_amdgcn_tanhf(x)
#else
#define TANH_(x) tanhf(x)
#endif

__device__ __forceinline__ float sigmoidf_(float x) {
    return 1.0f / (1.0f + __expf(-x));
}

__device__ __forceinline__ v8f zero8_() {
    v8f z = {0.f, 0.f, 0.f, 0.f, 0.f, 0.f, 0.f, 0.f};
    return z;
}

// Load one 16x32 f16 operand tile (row-major, leading dim `ld` halfs) into the
// CDNA5 WMMA 16-bit operand layout:
//   lane L (<16):  row L,    halfs = K[0..7]  then K[16..23]
//   lane L (>=16): row L-16, halfs = K[8..15] then K[24..31]
// Both 8-half chunks are 16-byte loads (offsets 0 and 32 bytes).
__device__ __forceinline__ v16h load_frag(const _Float16* __restrict__ p0, int ld) {
    const int lane = threadIdx.x & 31;
    const int row  = lane & 15;
    const int kb   = (lane >> 4) << 3;   // 0 or 8
    const _Float16* p = p0 + (size_t)row * ld + kb;
    union { v16h v; v8h h[2]; } u;
    u.h[0] = *(const v8h*)(p);
    u.h[1] = *(const v8h*)(p + 16);
    return u.v;
}

// ---------------------------------------------------------------------------
// Augmented-weight builders (f32 -> f16, packed for aligned WMMA B operands)
// ---------------------------------------------------------------------------

// Wenc[n][0:256]=enc_Whh, [256:268]=enc_Wih, [268]=enc_b, rest 0
__global__ void prep_enc_kernel(const float* __restrict__ Whh, const float* __restrict__ Wih,
                                const float* __restrict__ b, _Float16* __restrict__ dst) {
    int i = blockIdx.x * blockDim.x + threadIdx.x;
    if (i >= G_ * KE_) return;
    int n = i / KE_, c = i % KE_;
    float v = 0.f;
    if (c < 256)      v = Whh[(size_t)n * H_ + c];
    else if (c < 268) v = Wih[(size_t)n * IN_ + (c - 256)];
    else if (c == 268) v = b[n];
    dst[i] = (_Float16)v;
}

// Wattn[n][0:268]=attn_W row, [268]=attn_b, rest 0
__global__ void prep_attn_kernel(const float* __restrict__ W, const float* __restrict__ b,
                                 _Float16* __restrict__ dst) {
    int i = blockIdx.x * blockDim.x + threadIdx.x;
    if (i >= S_ * KE_) return;
    int n = i / KE_, c = i % KE_;
    float v = 0.f;
    if (c < 268)      v = W[(size_t)n * (H_ + IN_) + c];
    else if (c == 268) v = b[n];
    dst[i] = (_Float16)v;
}

// Wdec[n][0:268]=dec_Wih row ([din|attn]), [268:524]=dec_Whh, [524]=dec_b, rest 0
__global__ void prep_dec_kernel(const float* __restrict__ Wih, const float* __restrict__ Whh,
                                const float* __restrict__ b, _Float16* __restrict__ dst) {
    int i = blockIdx.x * blockDim.x + threadIdx.x;
    if (i >= G_ * KD_) return;
    int n = i / KD_, c = i % KD_;
    float v = 0.f;
    if (c < 268)      v = Wih[(size_t)n * (IN_ + H_) + c];
    else if (c < 524) v = Whh[(size_t)n * H_ + (c - 268)];
    else if (c == 524) v = b[n];
    dst[i] = (_Float16)v;
}

// Wlin1[n][0:256]=lin1_W, [268]=lin1_b, rest 0
__global__ void prep_lin1_kernel(const float* __restrict__ W, const float* __restrict__ b,
                                 _Float16* __restrict__ dst) {
    int i = blockIdx.x * blockDim.x + threadIdx.x;
    if (i >= H_ * KE_) return;
    int n = i / KE_, c = i % KE_;
    float v = 0.f;
    if (c < 256)      v = W[(size_t)n * H_ + c];
    else if (c == 268) v = b[n];
    dst[i] = (_Float16)v;
}

// ---------------------------------------------------------------------------
// Encoder: 16 WGs x 256 threads (8 waves). WG owns 16 batch rows, 512 steps.
// Per step: gates[16x1024] = [h|x_t|1] @ Wenc^T  (9 WMMAs per 16-col tile)
// Dynamic LDS: gates f32 [16][1024] @0      (65536)
//              cs    f32 [16][256]  @65536  (16384)
//              hx    f16 [16][288]  @81920  (9216)    [h | x_t | 1 | pad]
// ---------------------------------------------------------------------------
#define ENC_SMEM 91136

__global__ __launch_bounds__(256)
void encoder_kernel(const float* __restrict__ x,          // [B,S,IN]
                    const _Float16* __restrict__ Wenc,    // [G][KE_]
                    _Float16* __restrict__ enc_out_h,     // [B,S,H] f16
                    float* __restrict__ h_last) {         // [B,H] f32
    extern __shared__ char smem[];
    float*    gates = (float*)(smem);
    float*    cs    = (float*)(smem + 65536);
    _Float16* hx    = (_Float16*)(smem + 81920);

    const int tid  = threadIdx.x;
    const int lane = tid & 31;
    const int wave = tid >> 5;
    const int b0   = blockIdx.x * 16;
    const int nlo  = lane & 15;
    const int mhi  = (lane >> 4) * 8;

    for (int i = tid; i < 16 * KE_; i += 256) hx[i] = (_Float16)0.0f;
    for (int i = tid; i < 16 * H_;  i += 256) cs[i] = 0.0f;
    __syncthreads();
    if (tid < 16) hx[tid * KE_ + 268] = (_Float16)1.0f;   // bias lane
    __syncthreads();

    for (int t = 0; t < S_; ++t) {
        if (tid < 16 * IN_) {
            int m = tid / IN_, k = tid % IN_;
            hx[m * KE_ + 256 + k] = (_Float16)x[((size_t)(b0 + m) * S_ + t) * IN_ + k];
        }
        __syncthreads();

        // hoist loop-invariant A fragments (9 x 8 VGPRs, from LDS)
        v16h af[9];
#pragma unroll
        for (int kk = 0; kk < 9; ++kk) af[kk] = load_frag(hx + kk * 32, KE_);

#pragma clang loop unroll(disable)
        for (int nt = 0; nt < 8; ++nt) {
            const int n0 = wave * 128 + nt * 16;
            if (nt < 7)
                __builtin_prefetch(Wenc + (size_t)(n0 + 16) * KE_, 0, 1);
            v8f acc = zero8_();
#pragma unroll
            for (int kk = 0; kk < 9; ++kk) {
                v16h bf = load_frag(Wenc + (size_t)n0 * KE_ + kk * 32, KE_);
                acc = __builtin_amdgcn_wmma_f32_16x16x32_f16(
                    false, af[kk], false, bf, (short)0, acc, false, false);
            }
            const int n = n0 + nlo;
#pragma unroll
            for (int r = 0; r < 8; ++r) gates[(mhi + r) * G_ + n] = acc[r];
        }
        __syncthreads();

        // elementwise LSTM cell update (gate order: i, f, g, o)
#pragma unroll
        for (int m = 0; m < 16; ++m) {
            const int j  = tid;
            const float ig = gates[m * G_ + j];
            const float fg = gates[m * G_ + 256 + j];
            const float gg = gates[m * G_ + 512 + j];
            const float og = gates[m * G_ + 768 + j];
            float c = cs[m * H_ + j];
            c = sigmoidf_(fg) * c + sigmoidf_(ig) * TANH_(gg);
            const float h = sigmoidf_(og) * TANH_(c);
            cs[m * H_ + j] = c;
            hx[m * KE_ + j] = (_Float16)h;
            enc_out_h[((size_t)(b0 + m) * S_ + t) * H_ + j] = (_Float16)h;
            if (t == S_ - 1) h_last[(size_t)(b0 + m) * H_ + j] = h;
        }
        __syncthreads();
    }
}

// ---------------------------------------------------------------------------
// Decoder: 16 WGs x 256 threads, 64 sequential steps.
// Dynamic LDS: gates  f32 [16][1024] @0       (65536)
//              logits f32 [16][512]  @65536   (32768)
//              cs     f32 [16][256]  @98304   (16384)
//              hx     f16 [16][288]  @114688  (9216)   [h | din | 1 | pad]
//              abuf   f16 [16][544]  @123904  (17408)  [din | attn | h | 1 | pad]
//              rb     f16 [16][256]  @141312  (8192)   relu(lin1) out
//              red    f32 [256]      @149504  (1024)
// ---------------------------------------------------------------------------
#define DEC_SMEM 150528

__global__ __launch_bounds__(256)
void decoder_kernel(const float* __restrict__ target,      // [B,T,IN]
                    const float* __restrict__ lin2_W,      // [IN, H]
                    const float* __restrict__ lin2_b,      // [IN]
                    const _Float16* __restrict__ Wattn,    // [S][KE_]
                    const _Float16* __restrict__ Wdec,     // [G][KD_]
                    const _Float16* __restrict__ Wlin1,    // [H][KE_]
                    const _Float16* __restrict__ enc_out_h,// [B,S,H]
                    const float* __restrict__ h_last,      // [B,H]
                    float* __restrict__ out) {             // [B,T,IN]
    extern __shared__ char smem[];
    float*    gates  = (float*)(smem);
    float*    logits = (float*)(smem + 65536);
    float*    cs     = (float*)(smem + 98304);
    _Float16* hx     = (_Float16*)(smem + 114688);
    _Float16* abuf   = (_Float16*)(smem + 123904);
    _Float16* rb     = (_Float16*)(smem + 141312);
    float*    red    = (float*)(smem + 149504);

    const int tid  = threadIdx.x;
    const int lane = tid & 31;
    const int wave = tid >> 5;
    const int b0   = blockIdx.x * 16;
    const int nlo  = lane & 15;
    const int mhi  = (lane >> 4) * 8;

    // init: zero augmented buffers, then fill h, bias-ones, dec_in0
    for (int i = tid; i < 16 * KE_; i += 256) hx[i]   = (_Float16)0.0f;
    for (int i = tid; i < 16 * KD_; i += 256) abuf[i] = (_Float16)0.0f;
    __syncthreads();
    for (int i = tid; i < 16 * H_; i += 256) {
        int m = i >> 8, j = i & 255;
        float h = h_last[(size_t)b0 * H_ + i];
        hx[m * KE_ + j]         = (_Float16)h;
        abuf[m * KD_ + 268 + j] = (_Float16)h;
        cs[i] = 0.0f;
    }
    if (tid < 16) {
        hx[tid * KE_ + 268]   = (_Float16)1.0f;
        abuf[tid * KD_ + 524] = (_Float16)1.0f;
    }
    if (tid < 16 * IN_) {
        int m = tid / IN_, k = tid % IN_;
        float v = 0.0f;
        if (k == 10) v = 1.0f;
        if (k >= 4 && k < 10) v = target[((size_t)(b0 + m) * T_ + 0) * IN_ + k];
        hx[m * KE_ + 256 + k] = (_Float16)v;
        abuf[m * KD_ + k]     = (_Float16)v;
    }
    __syncthreads();

    for (int t = 0; t < T_; ++t) {
        // ---- Phase 1: logits = [h|din|1] @ Wattn^T  (N=512, 4 tiles/wave) ----
        {
            v16h af[9];
#pragma unroll
            for (int kk = 0; kk < 9; ++kk) af[kk] = load_frag(hx + kk * 32, KE_);
#pragma clang loop unroll(disable)
            for (int nt = 0; nt < 4; ++nt) {
                const int n0 = wave * 64 + nt * 16;
                if (nt < 3)
                    __builtin_prefetch(Wattn + (size_t)(n0 + 16) * KE_, 0, 1);
                v8f acc = zero8_();
#pragma unroll
                for (int kk = 0; kk < 9; ++kk) {
                    v16h bf = load_frag(Wattn + (size_t)n0 * KE_ + kk * 32, KE_);
                    acc = __builtin_amdgcn_wmma_f32_16x16x32_f16(
                        false, af[kk], false, bf, (short)0, acc, false, false);
                }
                const int n = n0 + nlo;
#pragma unroll
                for (int r = 0; r < 8; ++r) logits[(mhi + r) * S_ + n] = acc[r];
            }
        }
        __syncthreads();

        // ---- Phase 2: row-wise softmax over S=512 (16 threads per row) ----
        {
            const int r = tid >> 4, i = tid & 15;
            float mx = -1e30f;
            for (int s = i; s < S_; s += 16) mx = fmaxf(mx, logits[r * S_ + s]);
            red[tid] = mx;
            __syncthreads();
            if (i == 0) {
                float m2 = red[tid];
                for (int ii = 1; ii < 16; ++ii) m2 = fmaxf(m2, red[tid + ii]);
                red[tid] = m2;
            }
            __syncthreads();
            mx = red[r * 16];
            float sum = 0.0f;
            for (int s = i; s < S_; s += 16) {
                float e = __expf(logits[r * S_ + s] - mx);
                logits[r * S_ + s] = e;
                sum += e;
            }
            __syncthreads();
            red[tid] = sum;
            __syncthreads();
            if (i == 0) {
                float s2 = 0.0f;
                for (int ii = 0; ii < 16; ++ii) s2 += red[tid + ii];
                red[tid] = s2;
            }
            __syncthreads();
            const float inv = 1.0f / red[r * 16];
            for (int s = i; s < S_; s += 16) logits[r * S_ + s] *= inv;
        }
        __syncthreads();

        // ---- Phase 3: attn[m][j] = sum_s aw[m][s] * enc_out[b,s,j] ----
        {
            const int j = tid;   // 0..255
#pragma unroll 1
            for (int m = 0; m < 16; ++m) {
                const _Float16* ep = enc_out_h + (size_t)(b0 + m) * S_ * H_ + j;
                const float* aw = logits + m * S_;
                float a = 0.0f;
#pragma unroll 4
                for (int s = 0; s < S_; ++s) a += aw[s] * (float)ep[(size_t)s * H_];
                abuf[m * KD_ + 12 + j] = (_Float16)a;
            }
        }
        __syncthreads();

        // ---- Phase 4: gates = [din|attn|h|1] @ Wdec^T (N=1024, 8 tiles/wave) ----
        // A fragments reloaded from LDS per tile (17 frags held would spill).
#pragma clang loop unroll(disable)
        for (int nt = 0; nt < 8; ++nt) {
            const int n0 = wave * 128 + nt * 16;
            if (nt < 7)
                __builtin_prefetch(Wdec + (size_t)(n0 + 16) * KD_, 0, 1);
            v8f acc = zero8_();
#pragma unroll
            for (int kk = 0; kk < 17; ++kk) {
                v16h a  = load_frag(abuf + kk * 32, KD_);
                v16h bf = load_frag(Wdec + (size_t)n0 * KD_ + kk * 32, KD_);
                acc = __builtin_amdgcn_wmma_f32_16x16x32_f16(
                    false, a, false, bf, (short)0, acc, false, false);
            }
            const int n = n0 + nlo;
#pragma unroll
            for (int r = 0; r < 8; ++r) gates[(mhi + r) * G_ + n] = acc[r];
        }
        __syncthreads();

        // ---- elementwise LSTM cell update ----
#pragma unroll
        for (int m = 0; m < 16; ++m) {
            const int j  = tid;
            const float ig = gates[m * G_ + j];
            const float fg = gates[m * G_ + 256 + j];
            const float gg = gates[m * G_ + 512 + j];
            const float og = gates[m * G_ + 768 + j];
            float c = cs[m * H_ + j];
            c = sigmoidf_(fg) * c + sigmoidf_(ig) * TANH_(gg);
            const float h = sigmoidf_(og) * TANH_(c);
            cs[m * H_ + j] = c;
            hx[m * KE_ + j]         = (_Float16)h;
            abuf[m * KD_ + 268 + j] = (_Float16)h;
        }
        __syncthreads();

        // ---- Phase 5: relu([h|..|b] @ Wlin1^T) (N=256, 2 tiles/wave) ----
        // hx cols 256..267 hold din; Wlin1 has zeros there, bias at col 268.
        {
            v16h af[9];
#pragma unroll
            for (int kk = 0; kk < 9; ++kk) af[kk] = load_frag(hx + kk * 32, KE_);
#pragma clang loop unroll(disable)
            for (int nt = 0; nt < 2; ++nt) {
                const int n0 = wave * 32 + nt * 16;
                v8f acc = zero8_();
#pragma unroll
                for (int kk = 0; kk < 9; ++kk) {
                    v16h bf = load_frag(Wlin1 + (size_t)n0 * KE_ + kk * 32, KE_);
                    acc = __builtin_amdgcn_wmma_f32_16x16x32_f16(
                        false, af[kk], false, bf, (short)0, acc, false, false);
                }
                const int n = n0 + nlo;
#pragma unroll
                for (int r = 0; r < 8; ++r)
                    rb[(mhi + r) * H_ + n] = (_Float16)fmaxf(acc[r], 0.0f);
            }
        }
        __syncthreads();

        // ---- Phase 6: out = relu_out @ lin2_W.T + lin2_b ; feedback din ----
        if (tid < 16 * IN_) {
            const int m = tid / IN_, j = tid % IN_;
            float s = lin2_b[j];
            const float* w = lin2_W + (size_t)j * H_;
#pragma unroll 4
            for (int k = 0; k < H_; ++k) s += (float)rb[m * H_ + k] * w[k];
            out[((size_t)(b0 + m) * T_ + t) * IN_ + j] = s;
            hx[m * KE_ + 256 + j] = (_Float16)s;
            abuf[m * KD_ + j]     = (_Float16)s;
        }
        __syncthreads();
    }
}

// ---------------------------------------------------------------------------
// Host launcher
// ---------------------------------------------------------------------------
extern "C" void kernel_launch(void* const* d_in, const int* in_sizes, int n_in,
                              void* d_out, int out_size, void* d_ws, size_t ws_size,
                              hipStream_t stream) {
    const float* x        = (const float*)d_in[0];
    const float* target   = (const float*)d_in[1];
    const float* enc_Wih  = (const float*)d_in[2];
    const float* enc_Whh  = (const float*)d_in[3];
    const float* enc_b    = (const float*)d_in[4];
    const float* attn_W   = (const float*)d_in[5];
    const float* attn_b   = (const float*)d_in[6];
    const float* dec_Wih  = (const float*)d_in[7];
    const float* dec_Whh  = (const float*)d_in[8];
    const float* dec_b    = (const float*)d_in[9];
    const float* lin1_W   = (const float*)d_in[10];
    const float* lin1_b   = (const float*)d_in[11];
    const float* lin2_W   = (const float*)d_in[12];
    const float* lin2_b   = (const float*)d_in[13];
    float* out = (float*)d_out;

    // Workspace layout (bytes)
    char* ws = (char*)d_ws;
    _Float16* enc_out_h = (_Float16*)(ws);               // 256*512*256*2 = 67,108,864
    _Float16* Wenc      = (_Float16*)(ws + 67108864);    // 1024*288*2 =     589,824
    _Float16* Wattn     = (_Float16*)(ws + 67698688);    //  512*288*2 =     294,912
    _Float16* Wdec      = (_Float16*)(ws + 67993600);    // 1024*544*2 =   1,114,112
    _Float16* Wlin1     = (_Float16*)(ws + 69107712);    //  256*288*2 =     147,456
    float*    h_last    = (float*)(ws + 69255168);       //  256*256*4 =     262,144

    prep_enc_kernel <<<(G_ * KE_ + 255) / 256, 256, 0, stream>>>(enc_Whh, enc_Wih, enc_b, Wenc);
    prep_attn_kernel<<<(S_ * KE_ + 255) / 256, 256, 0, stream>>>(attn_W, attn_b, Wattn);
    prep_dec_kernel <<<(G_ * KD_ + 255) / 256, 256, 0, stream>>>(dec_Wih, dec_Whh, dec_b, Wdec);
    prep_lin1_kernel<<<(H_ * KE_ + 255) / 256, 256, 0, stream>>>(lin1_W, lin1_b, Wlin1);

    encoder_kernel<<<B_ / 16, 256, ENC_SMEM, stream>>>(x, Wenc, enc_out_h, h_last);

    decoder_kernel<<<B_ / 16, 256, DEC_SMEM, stream>>>(
        target, lin2_W, lin2_b, Wattn, Wdec, Wlin1, enc_out_h, h_last, out);
}
